// AWAttention_71734543778413
// MI455X (gfx1250) — matrix operand order repaired and checked by reading, not compiled, not run on hardware
//
#include <hip/hip_runtime.h>
#include <hip/hip_bf16.h>

// ---------------------------------------------------------------------------
// Problem sizes (fixed by the reference)
// ---------------------------------------------------------------------------
#define MDIM      8192
#define NDIM      8192
#define NFEAT     1024
#define MODELD    1024
#define HDIM      512

typedef __attribute__((ext_vector_type(16))) __bf16        v16bf;
typedef __attribute__((ext_vector_type(8)))  float         v8f;
typedef __attribute__((ext_vector_type(4)))  unsigned int  u32x4;
typedef __attribute__((ext_vector_type(8)))  unsigned int  u32x8;

union FragB16 {
    v16bf          v;
    unsigned short u[16];
    uint4          q[2];
};

// float -> bf16 with round-to-nearest-even
__device__ __forceinline__ unsigned short f2bf(float f) {
    unsigned int b = __builtin_bit_cast(unsigned int, f);
    b += 0x7FFFu + ((b >> 16) & 1u);
    return (unsigned short)(b >> 16);
}

// ---------------------------------------------------------------------------
// Tensor Data Mover: DMA a 2D bf16 tile (tile_rows x row_elems, row stride
// row_elems) from global memory into LDS at byte offset lds_off.
// D# layout per CDNA5 ISA ch.8: group0 = {count/flags, lds_addr, global_addr,
// type}, group1 = {mask/data_size, dims, tile dims, strides}.
// Issue from ONE wave only; completion via s_wait_tensorcnt.
// ---------------------------------------------------------------------------
__device__ __forceinline__ void tdm_load_tile_2d_bf16(unsigned lds_off,
                                                      const void* gptr,
                                                      unsigned row_elems,
                                                      unsigned tile_rows)
{
    const unsigned long long ga = (unsigned long long)gptr;
    u32x4 g0;
    g0[0] = 1u;                                                // count=1, user desc
    g0[1] = lds_off;                                           // lds_addr
    g0[2] = (unsigned)ga;                                      // global_addr[31:0]
    g0[3] = (unsigned)((ga >> 32) & 0x1FFFFFFu) | (2u << 30);  // addr[56:32] | type=2

    u32x8 g1;
    g1[0] = (1u << 16);                                        // wg_mask=0, data_size=1 (2B)
    g1[1] = (row_elems & 0xFFFFu) << 16;                       // tensor_dim0[15:0]
    g1[2] = ((row_elems >> 16) & 0xFFFFu) |
            ((tile_rows & 0xFFFFu) << 16);                     // dim0 hi | tensor_dim1 lo
    g1[3] = ((tile_rows >> 16) & 0xFFFFu) |
            ((row_elems & 0xFFFFu) << 16);                     // dim1 hi | tile_dim0
    g1[4] = (tile_rows & 0xFFFFu);                             // tile_dim1 (tile_dim2=0)
    g1[5] = row_elems;                                         // tensor_dim0_stride[31:0]
    g1[6] = 0u;                                                // stride0 hi | stride1 lo
    g1[7] = 0u;                                                // stride1 hi

    asm volatile("tensor_load_to_lds %0, %1" :: "s"(g0), "s"(g1) : "memory");
}

// ---------------------------------------------------------------------------
// Kernel 1: Out[m,n] = bf16( In[m,:] . W[n,:] + bias[n] )   (W stored [H,K])
// One wave per 16x16 output tile, K-loop of bf16 WMMAs.
// ---------------------------------------------------------------------------
__global__ void proj_bf16_kernel(const float* __restrict__ In,
                                 const float* __restrict__ W,
                                 const float* __restrict__ bias,
                                 unsigned short* __restrict__ Out,
                                 int Mrows, int Hdim, int Kdim)
{
    const int lane = threadIdx.x & 31;
    const int wave = threadIdx.x >> 5;
    const int h    = lane >> 4;     // half-wave
    const int l15  = lane & 15;

    const int tile    = blockIdx.x * (blockDim.x >> 5) + wave;
    const int tiles_n = Hdim >> 4;
    const int tm      = tile / tiles_n;
    const int tn      = tile - tm * tiles_n;
    if (tm * 16 >= Mrows) return;

    const int m = tm * 16 + l15;       // A-fragment row (lane = M)
    const int n = tn * 16 + l15;       // B-fragment col (lane = N)
    const float* __restrict__ arow = In + (size_t)m * Kdim;
    const float* __restrict__ wrow = W  + (size_t)n * Kdim;

    v8f acc = {};
    for (int kk = 0; kk < Kdim; kk += 32) {
        FragB16 fa, fb;
        // A 16x32 bf16 layout: elems 0..7 -> k = kk+h*8+j ; elems 8..15 -> k = kk+16+h*8+j
        #pragma unroll
        for (int j = 0; j < 8; ++j) {
            fa.u[j]     = f2bf(arow[kk + h * 8 + j]);
            fa.u[8 + j] = f2bf(arow[kk + 16 + h * 8 + j]);
        }
        // B 32x16 bf16 layout: elems 0..7 -> k = kk+h*8+j ; elems 8..15 -> k = kk+16+h*8+j, col = l15
        #pragma unroll
        for (int j = 0; j < 8; ++j) {
            fb.u[j]     = f2bf(wrow[kk + h * 8 + j]);
            fb.u[8 + j] = f2bf(wrow[kk + 16 + h * 8 + j]);
        }

        acc = __builtin_amdgcn_wmma_f32_16x16x32_bf16(false, fa.v, false, fb.v,
                                                      (short)0, acc, false, false);
    }

    const float b = bias[n];
    #pragma unroll
    for (int r = 0; r < 8; ++r) {
        // C/D layout: elem r, half h -> row M = r + 8*h, col N = l15
        const int row = tm * 16 + h * 8 + r;
        Out[(size_t)row * Hdim + n] = f2bf(acc[r] + b);
    }
}

// ---------------------------------------------------------------------------
// Kernel 2: elementwise f32 -> bf16
// ---------------------------------------------------------------------------
__global__ void f32_to_bf16_kernel(const float* __restrict__ in,
                                   unsigned short* __restrict__ out, int n4)
{
    const int stride = gridDim.x * blockDim.x;
    for (int i = blockIdx.x * blockDim.x + threadIdx.x; i < n4; i += stride) {
        float4 f = ((const float4*)in)[i];
        ushort4 o;
        o.x = f2bf(f.x); o.y = f2bf(f.y); o.z = f2bf(f.z); o.w = f2bf(f.w);
        ((ushort4*)out)[i] = o;
    }
}

// ---------------------------------------------------------------------------
// Kernel 3: flash attention  out = softmax(Q K^T) X
//   Q,K bf16 [8192,512], X bf16 [8192,1024], out f32 [8192,1024]
//   Workgroup: 8 waves, BLOCK_M = 32 rows, BLOCK_N = 32 cols per step.
//   Q/K tiles staged global->LDS by the Tensor Data Mover (one DMA per tile);
//   X tile staged transposed by the threads (byte reorder needs VGPRs).
//   Waves 0/1 own the softmax rows (row stats stay in-wave via shfl_xor
//   because the f32 C-layout keeps a row inside one 16-lane half).
//   Every wave owns a 128-wide slice of the 1024 output columns:
//   2 (row tiles) x 8 (col tiles) v8f accumulators = 128 VGPRs/lane.
// ---------------------------------------------------------------------------
#define QS_OFF  0
#define KS_OFF  (32 * 512 * 2)
#define FA_LDS_BYTES (32*512*2 /*Qs*/ + 32*512*2 /*Ks*/ + 1024*32*2 /*Xst*/ \
                      + 32*32*2 /*Ps*/ + 32*4 /*As*/ + 32*4 /*Ls*/)

__global__ void flash_attn_kernel(const unsigned short* __restrict__ Qb,
                                  const unsigned short* __restrict__ Kb,
                                  const unsigned short* __restrict__ Xb,
                                  float* __restrict__ Out)
{
    extern __shared__ char smem[];
    unsigned short* Qs  = (unsigned short*)smem;            // [32][512]
    unsigned short* Ks  = Qs  + 32 * 512;                   // [32][512]
    unsigned short* Xst = Ks  + 32 * 512;                   // [1024][32] (col-major X tile)
    unsigned short* Ps  = Xst + 1024 * 32;                  // [32][32]
    float*          As  = (float*)(Ps + 32 * 32);           // [32] rescale factors
    float*          Ls  = As + 32;                          // [32] softmax denominators

    const int tid  = threadIdx.x;
    const int lane = tid & 31;
    const int wave = tid >> 5;
    const int h    = lane >> 4;
    const int l15  = lane & 15;
    const int m0   = blockIdx.x * 32;

    // ---- stage Q tile via TDM: 32 rows x 512 bf16 ----
    if (wave == 0) {
        tdm_load_tile_2d_bf16(QS_OFF, Qb + (size_t)m0 * HDIM, HDIM, 32);
        __builtin_amdgcn_s_wait_tensorcnt(0);
    }

    v8f acc[2][8];
    #pragma unroll
    for (int t = 0; t < 2; ++t)
        #pragma unroll
        for (int j = 0; j < 8; ++j)
            acc[t][j] = (v8f){};

    float mrun[8], lrun[8];                // row stats (waves 0/1 only)
    #pragma unroll
    for (int r = 0; r < 8; ++r) { mrun[r] = -3.0e38f; lrun[r] = 0.0f; }

    const int c0 = wave * 128;             // this wave's output column base

    for (int n0 = 0; n0 < NDIM; n0 += 32) {
        __syncthreads();                   // prev O-update done before restaging

        // ---- stage K tile via TDM (wave 0 issues the DMA) ----
        if (wave == 0)
            tdm_load_tile_2d_bf16(KS_OFF, Kb + (size_t)n0 * HDIM, HDIM, 32);

        // ---- stage X tile transposed: Xst[col][row], 32 rows x 1024 cols ----
        for (int i = tid; i < 4096; i += 256) {
            const int r  = i >> 7;                         // source row 0..31
            const int cg = i & 127;                        // 8-col group
            uint4 d = ((const uint4*)(Xb + (size_t)(n0 + r) * MODELD))[cg];
            const unsigned short* e = (const unsigned short*)&d;
            const int cb = cg * 8;
            #pragma unroll
            for (int j = 0; j < 8; ++j)
                Xst[(cb + j) * 32 + r] = e[j];
        }
        // warm GL2 for next iteration's X tile
        if (n0 + 32 < NDIM) {
            const int pr = tid >> 3;                       // 32 rows over 256 threads
            const int pc = (tid & 7) * 128;
            __builtin_prefetch(Xb + (size_t)(n0 + 32 + pr) * MODELD + pc, 0, 1);
        }
        if (wave == 0)
            __builtin_amdgcn_s_wait_tensorcnt(0);
        __syncthreads();

        // ---- waves 0/1: S strip [16 x 32] + online softmax ----
        if (wave < 2) {
            v8f s0 = {}, s1 = {};
            const int mrow = wave * 16 + l15;
            for (int kk = 0; kk < HDIM; kk += 32) {
                FragB16 fa, fb0, fb1;
                fa.q[0]  = *(const uint4*)(Qs + mrow * 512 + kk + h * 8);
                fa.q[1]  = *(const uint4*)(Qs + mrow * 512 + kk + 16 + h * 8);
                // S = Q K^T : B(k, n) = K[n][k]  (Ks rows are N, contiguous in k)
                fb0.q[0] = *(const uint4*)(Ks + l15 * 512 + kk + h * 8);
                fb0.q[1] = *(const uint4*)(Ks + l15 * 512 + kk + 16 + h * 8);
                fb1.q[0] = *(const uint4*)(Ks + (16 + l15) * 512 + kk + h * 8);
                fb1.q[1] = *(const uint4*)(Ks + (16 + l15) * 512 + kk + 16 + h * 8);
                s0 = __builtin_amdgcn_wmma_f32_16x16x32_bf16(false, fa.v, false, fb0.v,
                                                             (short)0, s0, false, false);
                s1 = __builtin_amdgcn_wmma_f32_16x16x32_bf16(false, fa.v, false, fb1.v,
                                                             (short)0, s1, false, false);
            }
            #pragma unroll
            for (int r = 0; r < 8; ++r) {
                // row = wave*16 + h*8 + r ; cols: s0 -> l15, s1 -> 16+l15
                float v = fmaxf(s0[r], s1[r]);
                v = fmaxf(v, __shfl_xor(v, 8, 32));
                v = fmaxf(v, __shfl_xor(v, 4, 32));
                v = fmaxf(v, __shfl_xor(v, 2, 32));
                v = fmaxf(v, __shfl_xor(v, 1, 32));
                const float mnew  = fmaxf(mrun[r], v);
                const float alpha = __expf(mrun[r] - mnew);
                const float p0 = __expf(s0[r] - mnew);
                const float p1 = __expf(s1[r] - mnew);
                float ps = p0 + p1;
                ps += __shfl_xor(ps, 8, 32);
                ps += __shfl_xor(ps, 4, 32);
                ps += __shfl_xor(ps, 2, 32);
                ps += __shfl_xor(ps, 1, 32);
                lrun[r] = lrun[r] * alpha + ps;
                mrun[r] = mnew;
                const int row = wave * 16 + h * 8 + r;
                Ps[row * 32 + l15]      = f2bf(p0);
                Ps[row * 32 + 16 + l15] = f2bf(p1);
                if (l15 == 0) As[row] = alpha;
            }
        }
        __syncthreads();

        // ---- all waves: rescale O slice, accumulate P @ X_tile ----
        #pragma unroll
        for (int t = 0; t < 2; ++t) {
            FragB16 fp;
            const int prow = t * 16 + l15;
            fp.q[0] = *(const uint4*)(Ps + prow * 32 + h * 8);
            fp.q[1] = *(const uint4*)(Ps + prow * 32 + 16 + h * 8);
            #pragma unroll
            for (int r = 0; r < 8; ++r) {
                const float alpha = As[t * 16 + h * 8 + r];
                #pragma unroll
                for (int j = 0; j < 8; ++j) acc[t][j][r] *= alpha;
            }
            #pragma unroll
            for (int j = 0; j < 8; ++j) {
                FragB16 fx;
                const int col = c0 + j * 16 + l15;     // B col = lane
                fx.q[0] = *(const uint4*)(Xst + col * 32 + h * 8);
                fx.q[1] = *(const uint4*)(Xst + col * 32 + 16 + h * 8);
                acc[t][j] = __builtin_amdgcn_wmma_f32_16x16x32_bf16(false, fp.v, false, fx.v,
                                                                    (short)0, acc[t][j],
                                                                    false, false);
            }
        }
    }

    // ---- finalize: divide by softmax denominator, write f32 ----
    if (wave < 2) {
        #pragma unroll
        for (int r = 0; r < 8; ++r)
            if (l15 == 0) Ls[wave * 16 + h * 8 + r] = lrun[r];
    }
    __syncthreads();
    #pragma unroll
    for (int t = 0; t < 2; ++t) {
        #pragma unroll
        for (int r = 0; r < 8; ++r) {
            const float linv = 1.0f / Ls[t * 16 + h * 8 + r];
            const size_t row = (size_t)(m0 + t * 16 + h * 8 + r);
            #pragma unroll
            for (int j = 0; j < 8; ++j)
                Out[row * MODELD + c0 + j * 16 + l15] = acc[t][j][r] * linv;
        }
    }
}

// ---------------------------------------------------------------------------
// Launch
// ---------------------------------------------------------------------------
extern "C" void kernel_launch(void* const* d_in, const int* in_sizes, int n_in,
                              void* d_out, int out_size, void* d_ws, size_t ws_size,
                              hipStream_t stream)
{
    const float* A  = (const float*)d_in[0];   // [8192,1024]
    const float* X  = (const float*)d_in[1];   // [8192,1024]
    const float* Wq = (const float*)d_in[2];   // [512,1024]
    const float* bq = (const float*)d_in[3];   // [512]
    const float* Wk = (const float*)d_in[4];   // [512,1024]
    const float* bk = (const float*)d_in[5];   // [512]
    float*       out = (float*)d_out;          // [8192,1024]

    // workspace carve: Qb(8MB) | Kb(8MB) | Xb(16MB), all bf16
    unsigned short* Qb = (unsigned short*)d_ws;
    unsigned short* Kb = Qb + (size_t)MDIM * HDIM;
    unsigned short* Xb = Kb + (size_t)NDIM * HDIM;

    // Q = A Wq^T + bq ; K = X Wk^T + bk   (16384 tiles, 8 waves/block)
    proj_bf16_kernel<<<2048, 256, 0, stream>>>(A, Wq, bq, Qb, MDIM, HDIM, NFEAT);
    proj_bf16_kernel<<<2048, 256, 0, stream>>>(X, Wk, bk, Kb, NDIM, HDIM, MODELD);

    // X -> bf16 (V operand)
    f32_to_bf16_kernel<<<1024, 256, 0, stream>>>(X, Xb, (NDIM * MODELD) / 4);

    // fused softmax(QK^T) X, 32 M-rows per workgroup
    hipFuncSetAttribute((const void*)flash_attn_kernel,
                        hipFuncAttributeMaxDynamicSharedMemorySize, FA_LDS_BYTES);
    flash_attn_kernel<<<MDIM / 32, 256, FA_LDS_BYTES, stream>>>(Qb, Kb, Xb, out);
}
